// GraphModule_32719060861136
// MI455X (gfx1250) — compile-verified, matrix operand order checked by reading, not run on hardware
//
#include <hip/hip_runtime.h>
#include <hip/hip_bf16.h>
#include <stddef.h>

#define N_NODES 50000
#define N_EDGES 800000
#define IN_DIM  10
#define HID_DIM 384

typedef __attribute__((ext_vector_type(16))) __bf16 v16bf;
typedef __attribute__((ext_vector_type(8)))  float  v8f;
typedef __attribute__((ext_vector_type(4)))  float  v4f;

// -------------------------------------------------------------------------
// Pre-pack weight matrix B[K,N] (fp32, row-major) into bf16 WMMA B-fragments.
// Fragment layout (CDNA5 ISA 7.12.2, wave32, 16-bit B 32x16):
//   lane(0-15): col=lane,    K-offsets {0..7, 16..23}
//   lane(16-31): col=lane-16, K-offsets {8..15, 24..31}
// Buffer index: ((slab*nTiles + ntile)*32 + lane)*16 + e, zero-filled past K.
// One thread writes one lane's 16-element fragment (32 B, two b128 stores).
// -------------------------------------------------------------------------
__global__ void pack_b(const float* __restrict__ B, __bf16* __restrict__ Bp,
                       int N, int K)
{
    const int nTiles = N / 16;
    const int nSlabs = (K + 31) / 32;
    const int total  = nSlabs * nTiles * 32;
    int tid = blockIdx.x * blockDim.x + threadIdx.x;
    if (tid >= total) return;

    const int lane  = tid & 31;
    const int idx   = tid >> 5;
    const int ntile = idx % nTiles;
    const int slab  = idx / nTiles;
    const int half  = lane >> 4;
    const int col   = ntile * 16 + (lane & 15);

    v16bf frag;
#pragma unroll
    for (int e = 0; e < 16; ++e) {
        const int k = slab * 32 + half * 8 + e + ((e >= 8) ? 8 : 0);
        const int kc = (k < K) ? k : 0;                 // clamp: no branchy load
        float bv = B[(size_t)kc * N + col];
        frag[e] = (k < K) ? (__bf16)bv : (__bf16)0.0f;
    }
    *(v16bf*)(Bp + (size_t)tid * 16) = frag;
}

// -------------------------------------------------------------------------
// WMMA GEMM: C[M,N] = op(A)[M,K] @ B[K,N], A fp32 row-major, B pre-packed
// bf16 fragments, fp32 accumulate.  op = relu when RELU (fuses layer-1 ReLU).
// One wave -> one 16x16 C tile; 8 waves/block -> 128 rows/block.
// Requires M % 16 == 0 and N % 16 == 0 (true here: 50000, 384).
// Main loop: K full 32-slabs, unconditional b128 A loads + b128 B-frag loads.
// Tail (K % 32 != 0, i.e. layer-1 K=10): clamped-index loads + select.
// -------------------------------------------------------------------------
template <bool RELU>
__global__ __launch_bounds__(256) void gemm_bf16_wmma(
    const float* __restrict__ A, const __bf16* __restrict__ Bp,
    float* __restrict__ C, int M, int N, int K)
{
    const int lane = threadIdx.x & 31;
    const int wave = threadIdx.x >> 5;
    const int half = lane >> 4;
    const int r16  = lane & 15;

    const int tileM = (blockIdx.x * 8 + wave) * 16;
    const int tileN = blockIdx.y * 16;
    const int nTiles = N / 16;
    if (tileM >= M) return;

    const float* __restrict__ arow = A + (size_t)(tileM + r16) * K;
    const v16bf* __restrict__ bp =
        (const v16bf*)Bp + (size_t)blockIdx.y * 32 + lane;
    const int bstride = nTiles * 32;   // v16bf elements per slab
    const int kbase = half * 8;

    const int kFull  = K / 32;
    const int nSlabs = (K + 31) / 32;

    v8f acc = {};
    for (int s = 0; s < kFull; ++s) {
        const float* p = arow + s * 32 + kbase;
        v4f f0 = *(const v4f*)(p);
        v4f f1 = *(const v4f*)(p + 4);
        v4f f2 = *(const v4f*)(p + 16);
        v4f f3 = *(const v4f*)(p + 20);
        if (RELU) {
#pragma unroll
            for (int j = 0; j < 4; ++j) {
                f0[j] = fmaxf(f0[j], 0.0f); f1[j] = fmaxf(f1[j], 0.0f);
                f2[j] = fmaxf(f2[j], 0.0f); f3[j] = fmaxf(f3[j], 0.0f);
            }
        }
        v16bf a;
#pragma unroll
        for (int j = 0; j < 4; ++j) {
            a[j]      = (__bf16)f0[j];
            a[4 + j]  = (__bf16)f1[j];
            a[8 + j]  = (__bf16)f2[j];
            a[12 + j] = (__bf16)f3[j];
        }
        v16bf b = bp[(size_t)s * bstride];
        acc = __builtin_amdgcn_wmma_f32_16x16x32_bf16(
                  false, a, false, b, (short)0, acc, false, false);
    }

    if (kFull < nSlabs) {              // K remainder slab (layer 1: K=10)
        const int k0 = kFull * 32;
        v16bf a;
#pragma unroll
        for (int e = 0; e < 16; ++e) {
            const int k = k0 + kbase + e + ((e >= 8) ? 8 : 0);
            const int kc = (k < K) ? k : 0;        // clamped, unconditional load
            float av = arow[kc];
            if (RELU) av = fmaxf(av, 0.0f);
            a[e] = (k < K) ? (__bf16)av : (__bf16)0.0f;
        }
        v16bf b = bp[(size_t)kFull * bstride];
        acc = __builtin_amdgcn_wmma_f32_16x16x32_bf16(
                  false, a, false, b, (short)0, acc, false, false);
    }

    float* __restrict__ crow = C + (size_t)(tileM + half * 8) * N + tileN + r16;
#pragma unroll
    for (int v = 0; v < 8; ++v)
        crow[(size_t)v * N] = acc[v];
}

// ------------------------ degree / dinv pipeline -------------------------
__global__ void deg_init(float* __restrict__ deg) {
    int i = blockIdx.x * blockDim.x + threadIdx.x;
    if (i < N_NODES) deg[i] = 1.0f;   // self loop contributes 1
}

__global__ void deg_accum(const int* __restrict__ col, float* __restrict__ deg) {
    int i = blockIdx.x * blockDim.x + threadIdx.x;
    if (i < N_EDGES) atomicAdd(&deg[col[i]], 1.0f);
}

__global__ void deg_rsqrt(float* __restrict__ deg) {
    int i = blockIdx.x * blockDim.x + threadIdx.x;
    if (i < N_NODES) deg[i] = rsqrtf(deg[i]);
}

// out[i,f] = h[i,f]*dinv[i]^2 + bias[f]  (self-loop message + bias; also the
// deterministic initializer of the scatter destination)
__global__ void selfloop_bias_init(const float* __restrict__ h,
                                   const float* __restrict__ dinv,
                                   const float* __restrict__ bias,
                                   float* __restrict__ out, int total)
{
    int idx = blockIdx.x * blockDim.x + threadIdx.x;
    if (idx >= total) return;
    int node = idx / HID_DIM;
    int f    = idx - node * HID_DIM;
    float di = dinv[node];
    out[idx] = h[idx] * di * di + bias[f];
}

// One wave per edge: gather h[row]*norm, scatter-add into out[col].
// h/out are L2-resident (76.8 MB each < 192 MB L2): atomics resolve in L2.
__global__ __launch_bounds__(256) void edge_scatter(
    const int* __restrict__ row, const int* __restrict__ col,
    const float* __restrict__ dinv, const float* __restrict__ h,
    float* __restrict__ out)
{
    const int gid  = blockIdx.x * blockDim.x + threadIdx.x;
    const int lane = gid & 31;
    const int wav  = gid >> 5;
    const int nwav = (gridDim.x * blockDim.x) >> 5;

    for (int e = wav; e < N_EDGES; e += nwav) {
        const int r = row[e];
        const int c = col[e];
        const float norm = dinv[r] * dinv[c];
        const float* hr = h   + (size_t)r * HID_DIM;
        float*       oc = out + (size_t)c * HID_DIM;
#pragma unroll
        for (int f = lane; f < HID_DIM; f += 32)
            atomicAdd(&oc[f], hr[f] * norm);
    }
}

// -------------------------------------------------------------------------
extern "C" void kernel_launch(void* const* d_in, const int* in_sizes, int n_in,
                              void* d_out, int out_size, void* d_ws, size_t ws_size,
                              hipStream_t stream)
{
    const float* x  = (const float*)d_in[0];
    const int*   ei = (const int*)  d_in[1];   // [2, E] flat
    const float* W1 = (const float*)d_in[2];
    const float* b1 = (const float*)d_in[3];
    const float* W2 = (const float*)d_in[4];
    const float* b2 = (const float*)d_in[5];
    float* out = (float*)d_out;

    const int* row = ei;
    const int* col = ei + N_EDGES;

    // ws layout: dinv | Bp1 | Bp2 | bufA | bufB
    char* ws = (char*)d_ws;
    float*  dinv = (float*)ws;                                  // 200 KB
    __bf16* Bp1  = (__bf16*)(ws + (1u << 20));                  //  24 KB packed W1
    __bf16* Bp2  = (__bf16*)(ws + (1u << 20) + (1u << 19));     // 295 KB packed W2
    float*  bufA = (float*)(ws + (2u << 20));
    float*  bufB = (float*)(ws + (2u << 20) + (size_t)N_NODES * HID_DIM * sizeof(float));

    const int T = 256;
    const int total = N_NODES * HID_DIM;
    const int nTiles = HID_DIM / 16;                            // 24

    // degrees -> dinv
    deg_init <<<(N_NODES + T - 1) / T, T, 0, stream>>>(dinv);
    deg_accum<<<(N_EDGES + T - 1) / T, T, 0, stream>>>(col, dinv);
    deg_rsqrt<<<(N_NODES + T - 1) / T, T, 0, stream>>>(dinv);

    // pack weights into bf16 WMMA B-fragments
    {
        int tot1 = ((IN_DIM + 31) / 32) * nTiles * 32;          // 768
        int tot2 = ((HID_DIM + 31) / 32) * nTiles * 32;         // 9216
        pack_b<<<(tot1 + T - 1) / T, T, 0, stream>>>(W1, Bp1, HID_DIM, IN_DIM);
        pack_b<<<(tot2 + T - 1) / T, T, 0, stream>>>(W2, Bp2, HID_DIM, HID_DIM);
    }

    dim3 gemm_grid((N_NODES + 127) / 128, nTiles);

    // Layer 1: h1 = x @ W1 -> bufA
    gemm_bf16_wmma<false><<<gemm_grid, 256, 0, stream>>>(x, Bp1, bufA,
                                                         N_NODES, HID_DIM, IN_DIM);
    // t1 = Dhat^-1/2 (A+I) Dhat^-1/2 h1 + b1 -> bufB
    selfloop_bias_init<<<(total + T - 1) / T, T, 0, stream>>>(bufA, dinv, b1, bufB, total);
    edge_scatter<<<4096, 256, 0, stream>>>(row, col, dinv, bufA, bufB);

    // Layer 2: h2 = relu(t1) @ W2 -> bufA  (ReLU fused into A-fragment load)
    gemm_bf16_wmma<true><<<gemm_grid, 256, 0, stream>>>(bufB, Bp2, bufA,
                                                        N_NODES, HID_DIM, HID_DIM);
    // out = Dhat^-1/2 (A+I) Dhat^-1/2 h2 + b2 -> d_out
    selfloop_bias_init<<<(total + T - 1) / T, T, 0, stream>>>(bufA, dinv, b2, out, total);
    edge_scatter<<<4096, 256, 0, stream>>>(row, col, dinv, bufA, out);
}
